// Decoder_85289460564608
// MI455X (gfx1250) — compile-verified, matrix operand order checked by reading, not run on hardware
//
#include <hip/hip_runtime.h>
#include <hip/hip_bf16.h>
#include <math.h>

#define VOCAB 50257
#define EMBD  512
#define HH    2048
#define SS    4096
#define ENCD  2048

typedef __attribute__((ext_vector_type(16))) __bf16 v16bf;
typedef __attribute__((ext_vector_type(8)))  __bf16 v8bf;
typedef __attribute__((ext_vector_type(8)))  float  v8f;

__device__ __forceinline__ float sigmoid_(float x) { return 1.0f / (1.0f + __expf(-x)); }

// ---------------------------------------------------------------------------
// qb[j] = sum_i hidden[i]*W1[i,j] + W1_b[j] + W2_b[j]
// ---------------------------------------------------------------------------
__global__ void qw1_kernel(const float* __restrict__ hidden,
                           const float* __restrict__ W1,
                           const float* __restrict__ W1b,
                           const float* __restrict__ W2b,
                           float* __restrict__ qb) {
  int j = blockIdx.x * blockDim.x + threadIdx.x;
  if (j >= HH) return;
  float acc = 0.f;
  for (int i = 0; i < HH; ++i)
    acc = fmaf(hidden[i], W1[(size_t)i * HH + j], acc);
  qb[j] = acc + W1b[j] + W2b[j];
}

// ---------------------------------------------------------------------------
// Pack W2 (fp32, KxN row-major) into WMMA-ready bf16 tiles:
//   Bpk[((ntile*64 + ktile)*32 + lane)*16 + j] = bf16(W2[k, n])
//     n = ntile*16 + (lane&15)
//     k = ktile*32 + (lane>>4)*8 + (j&7) + (j>=8 ? 16 : 0)
// Each wave's B tile becomes one contiguous 1KB block -> 2x b128 per lane.
// Transpose goes through LDS so global reads and writes stay coalesced.
// grid = (128 ntiles, 16 kgroups of 128 rows), 256 threads.
// ---------------------------------------------------------------------------
__global__ void w2pack_kernel(const float* __restrict__ W2,
                              __bf16* __restrict__ Bpk) {
  __shared__ float tile[128 * 16];
  const int tid = threadIdx.x;
  const int n0 = blockIdx.x * 16;
  const int k0 = blockIdx.y * 128;
  // load 128 rows x 16 cols of W2 (row-major, cols fast -> coalesced 64B runs)
  for (int t = tid; t < 128 * 16; t += 256) {
    int r = t >> 4, c = t & 15;
    tile[t] = W2[(size_t)(k0 + r) * HH + (n0 + c)];
  }
  __syncthreads();
  // emit 4 packed ktiles (each 512 bf16), output fully coalesced
  for (int t = tid; t < 4 * 512; t += 256) {
    int kt   = t >> 9;         // ktile within group
    int rem  = t & 511;
    int lane = rem >> 4;
    int j    = rem & 15;
    int kloc = kt * 32 + ((lane >> 4) * 8) + (j & 7) + ((j >> 3) << 4);
    int nn   = lane & 15;
    size_t ktg = (size_t)blockIdx.y * 4 + kt;
    Bpk[(((size_t)blockIdx.x * 64 + ktg) * 32 + lane) * 16 + j] =
        (__bf16)tile[kloc * 16 + nn];
  }
}

// ---------------------------------------------------------------------------
// score[s] = sum_n tanh( (enc[s,:] @ W2)[n] + qb[n] ) * Vk[n] + Vb
// Packed-B version: inner loop = 2x ds_load_b128 + 2x global_load_b128 + wmma.
// ---------------------------------------------------------------------------
__launch_bounds__(128)
__global__ void attn_score_packed_kernel(const float* __restrict__ enc,
                                         const __bf16* __restrict__ Bpk,
                                         const float* __restrict__ qb,
                                         const float* __restrict__ Vk,
                                         const float* __restrict__ Vb,
                                         float* __restrict__ score) {
  __shared__ __align__(16) __bf16 Atile[16 * HH];   // 64 KB
  const int tid  = threadIdx.x;
  const int lane = tid & 31;
  const int wave = tid >> 5;
  const int s0   = blockIdx.x * 16;

  for (int t = tid; t < 16 * HH; t += 128) {
    int r = t >> 11, k = t & (HH - 1);
    Atile[t] = (__bf16)enc[(size_t)(s0 + r) * HH + k];
  }
  __syncthreads();

  const int m    = lane & 15;
  const int half = lane >> 4;
  const int kh   = half * 8;
  const int ncol = lane & 15;
  const __bf16* arow = &Atile[m * HH];

  float part[8];
#pragma unroll
  for (int r = 0; r < 8; ++r) part[r] = 0.f;

  for (int nt = 0; nt < 32; ++nt) {
    const int ntile = wave * 32 + nt;
    const __bf16* bptr = Bpk + ((size_t)ntile * 64) * 512 + lane * 16;
    v8f c = {};
#pragma unroll 4
    for (int kt = 0; kt < 64; ++kt) {
      const int k0 = kt * 32;
      v8bf alo = *(const v8bf*)(arow + k0 + kh);
      v8bf ahi = *(const v8bf*)(arow + k0 + kh + 16);
      v16bf a = __builtin_shufflevector(alo, ahi,
                                        0, 1, 2, 3, 4, 5, 6, 7,
                                        8, 9, 10, 11, 12, 13, 14, 15);
      v16bf b = *(const v16bf*)bptr;          // 32 contiguous bytes / lane
      bptr += 512;                            // next 1KB wave tile
      c = __builtin_amdgcn_wmma_f32_16x16x32_bf16(
              false, a, false, b, (short)0, c, false, false);
    }
    const int n0c = ntile * 16 + ncol;
    const float qv = qb[n0c];
    const float vv = Vk[n0c];
#pragma unroll
    for (int r = 0; r < 8; ++r)
      part[r] += tanhf(c[r] + qv) * vv;
  }

  __syncthreads();
  float* rowsum = (float*)Atile;
  if (tid < 16) rowsum[tid] = 0.f;
  __syncthreads();
#pragma unroll
  for (int r = 0; r < 8; ++r)
    atomicAdd(&rowsum[half * 8 + r], part[r]);
  __syncthreads();
  if (tid < 16) score[s0 + tid] = rowsum[tid] + Vb[0];
}

// ---------------------------------------------------------------------------
// Fallback (no workspace for packed B): direct strided loads of W2.
// ---------------------------------------------------------------------------
__launch_bounds__(128)
__global__ void attn_score_direct_kernel(const float* __restrict__ enc,
                                         const float* __restrict__ W2,
                                         const float* __restrict__ qb,
                                         const float* __restrict__ Vk,
                                         const float* __restrict__ Vb,
                                         float* __restrict__ score) {
  __shared__ __align__(16) __bf16 Atile[16 * HH];
  const int tid  = threadIdx.x;
  const int lane = tid & 31;
  const int wave = tid >> 5;
  const int s0   = blockIdx.x * 16;

  for (int t = tid; t < 16 * HH; t += 128) {
    int r = t >> 11, k = t & (HH - 1);
    Atile[t] = (__bf16)enc[(size_t)(s0 + r) * HH + k];
  }
  __syncthreads();

  const int m    = lane & 15;
  const int half = lane >> 4;
  const int kh   = half * 8;
  const int ncol = lane & 15;
  const __bf16* arow = &Atile[m * HH];

  float part[8];
#pragma unroll
  for (int r = 0; r < 8; ++r) part[r] = 0.f;

  for (int nt = 0; nt < 32; ++nt) {
    const int n0 = wave * 512 + nt * 16;
    const float* bcol = W2 + (n0 + ncol);
    v8f c = {};
    for (int k0 = 0; k0 < HH; k0 += 32) {
      v8bf alo = *(const v8bf*)(arow + k0 + kh);
      v8bf ahi = *(const v8bf*)(arow + k0 + kh + 16);
      v16bf a = __builtin_shufflevector(alo, ahi,
                                        0, 1, 2, 3, 4, 5, 6, 7,
                                        8, 9, 10, 11, 12, 13, 14, 15);
      v16bf b;
#pragma unroll
      for (int j = 0; j < 8; ++j) {
        b[j]     = (__bf16)bcol[(size_t)(k0 + kh + j) * HH];
        b[j + 8] = (__bf16)bcol[(size_t)(k0 + kh + 16 + j) * HH];
      }
      if (k0 + 32 < HH)
        __builtin_prefetch(&bcol[(size_t)(k0 + 32 + kh) * HH], 0, 1);
      c = __builtin_amdgcn_wmma_f32_16x16x32_bf16(
              false, a, false, b, (short)0, c, false, false);
    }
    const float qv = qb[n0 + ncol];
    const float vv = Vk[n0 + ncol];
#pragma unroll
    for (int r = 0; r < 8; ++r)
      part[r] += tanhf(c[r] + qv) * vv;
  }

  __syncthreads();
  float* rowsum = (float*)Atile;
  if (tid < 16) rowsum[tid] = 0.f;
  __syncthreads();
#pragma unroll
  for (int r = 0; r < 8; ++r)
    atomicAdd(&rowsum[half * 8 + r], part[r]);
  __syncthreads();
  if (tid < 16) score[s0 + tid] = rowsum[tid] + Vb[0];
}

// ---------------------------------------------------------------------------
__global__ void softmax_kernel(const float* __restrict__ score,
                               float* __restrict__ aw) {
  __shared__ float buf[256];
  const int tid = threadIdx.x;
  float m = -3.4e38f;
  for (int s = tid; s < SS; s += 256) m = fmaxf(m, score[s]);
  buf[tid] = m; __syncthreads();
  for (int o = 128; o > 0; o >>= 1) {
    if (tid < o) buf[tid] = fmaxf(buf[tid], buf[tid + o]);
    __syncthreads();
  }
  const float mx = buf[0];
  __syncthreads();
  float sum = 0.f;
  for (int s = tid; s < SS; s += 256) sum += expf(score[s] - mx);
  buf[tid] = sum; __syncthreads();
  for (int o = 128; o > 0; o >>= 1) {
    if (tid < o) buf[tid] += buf[tid + o];
    __syncthreads();
  }
  const float inv = 1.0f / buf[0];
  for (int s = tid; s < SS; s += 256) aw[s] = expf(score[s] - mx) * inv;
}

// ---------------------------------------------------------------------------
__global__ void ctx_kernel(const float* __restrict__ aw,
                           const float* __restrict__ enc,
                           float* __restrict__ ctx) {
  int j = blockIdx.x * blockDim.x + threadIdx.x;
  if (j >= HH) return;
  float acc = 0.f;
  for (int s = 0; s < SS; ++s)
    acc = fmaf(aw[s], enc[(size_t)s * HH + j], acc);
  ctx[j] = acc;
}

// ---------------------------------------------------------------------------
// Degenerate LSTM step (h0 = c0 = 0): forget gate & R matrix are dead.
// ---------------------------------------------------------------------------
__global__ void lstm_gemv_kernel(const float* __restrict__ xa, int Da,
                                 const float* __restrict__ xb, int Db,
                                 const int* __restrict__ idx,
                                 const float* __restrict__ K,
                                 const float* __restrict__ bias,
                                 float* __restrict__ hout,
                                 float* __restrict__ hout2) {
  int j = blockIdx.x * blockDim.x + threadIdx.x;
  if (j >= HH) return;
  const float* xbp = idx ? (xb + (size_t)idx[0] * Db) : xb;
  float a0 = 0.f, a2 = 0.f, a3 = 0.f;
  const size_t ld = 4 * HH;
  for (int i = 0; i < Da; ++i) {
    const float x = xa[i];
    const float* row = K + (size_t)i * ld;
    a0 = fmaf(x, row[j], a0);
    a2 = fmaf(x, row[2 * HH + j], a2);
    a3 = fmaf(x, row[3 * HH + j], a3);
  }
  const float* Kb = K + (size_t)Da * ld;
  for (int i = 0; i < Db; ++i) {
    const float x = xbp[i];
    const float* row = Kb + (size_t)i * ld;
    a0 = fmaf(x, row[j], a0);
    a2 = fmaf(x, row[2 * HH + j], a2);
    a3 = fmaf(x, row[3 * HH + j], a3);
  }
  const float zi = a0 + bias[j];
  const float zg = a2 + bias[2 * HH + j];
  const float zo = a3 + bias[3 * HH + j];
  const float c = sigmoid_(zi) * tanhf(zg);
  const float h = sigmoid_(zo) * tanhf(c);
  hout[j] = h;
  if (hout2) hout2[j] = h;
}

// ---------------------------------------------------------------------------
__global__ void logits_kernel(const float* __restrict__ h4,
                              const float* __restrict__ fck,
                              const float* __restrict__ fcb,
                              float* __restrict__ out) {
  int v = blockIdx.x * blockDim.x + threadIdx.x;
  if (v >= VOCAB) return;
  float acc = 0.f;
  for (int i = 0; i < HH; ++i)
    acc = fmaf(h4[i], fck[(size_t)i * VOCAB + v], acc);
  out[v] = acc + fcb[v];
}

// ---------------------------------------------------------------------------
extern "C" void kernel_launch(void* const* d_in, const int* in_sizes, int n_in,
                              void* d_out, int out_size, void* d_ws, size_t ws_size,
                              hipStream_t stream) {
  const int*   x      = (const int*)  d_in[0];
  const float* hidden = (const float*)d_in[1];
  const float* enc    = (const float*)d_in[2];
  const float* W1k    = (const float*)d_in[3];
  const float* W1b    = (const float*)d_in[4];
  const float* W2k    = (const float*)d_in[5];
  const float* W2b    = (const float*)d_in[6];
  const float* Vk     = (const float*)d_in[7];
  const float* Vb     = (const float*)d_in[8];
  const float* emb    = (const float*)d_in[9];
  const float* K1     = (const float*)d_in[10];
  const float* b1     = (const float*)d_in[12];
  const float* K2     = (const float*)d_in[13];
  const float* b2     = (const float*)d_in[15];
  const float* K3     = (const float*)d_in[16];
  const float* b3     = (const float*)d_in[18];
  const float* K4     = (const float*)d_in[19];
  const float* b4     = (const float*)d_in[21];
  const float* fck    = (const float*)d_in[22];
  const float* fcb    = (const float*)d_in[23];

  // Output layout: [logits 50257 | h4 2048 | aw 4096]
  float* out_logits = (float*)d_out;
  float* out_h4     = out_logits + VOCAB;
  float* out_aw     = out_h4 + HH;

  float* ws    = (float*)d_ws;
  float* qb    = ws;            // 2048
  float* score = qb + HH;       // 4096
  float* ctx   = score + SS;    // 2048
  float* h1    = ctx + HH;      // 2048
  float* h2    = h1 + HH;       // 2048
  float* h3    = h2 + HH;       // 2048
  float* h4    = h3 + HH;       // 2048
  // packed B lives after the float region (offset 64KB, well aligned)
  const size_t float_region = 16384 * sizeof(float);
  __bf16* Bpk = (__bf16*)((char*)d_ws + float_region);
  const bool use_packed =
      ws_size >= float_region + (size_t)HH * HH * sizeof(__bf16);

  qw1_kernel<<<HH / 256, 256, 0, stream>>>(hidden, W1k, W1b, W2b, qb);
  if (use_packed) {
    w2pack_kernel<<<dim3(HH / 16, HH / 128), 256, 0, stream>>>(W2k, Bpk);
    attn_score_packed_kernel<<<SS / 16, 128, 0, stream>>>(enc, Bpk, qb, Vk, Vb, score);
  } else {
    attn_score_direct_kernel<<<SS / 16, 128, 0, stream>>>(enc, W2k, qb, Vk, Vb, score);
  }
  softmax_kernel<<<1, 256, 0, stream>>>(score, out_aw);
  ctx_kernel<<<HH / 256, 256, 0, stream>>>(out_aw, enc, ctx);
  lstm_gemv_kernel<<<HH / 256, 256, 0, stream>>>(ctx, ENCD, emb, EMBD, x,  K1, b1, h1, nullptr);
  lstm_gemv_kernel<<<HH / 256, 256, 0, stream>>>(ctx, ENCD, h1, HH, nullptr, K2, b2, h2, nullptr);
  lstm_gemv_kernel<<<HH / 256, 256, 0, stream>>>(ctx, ENCD, h2, HH, nullptr, K3, b3, h3, nullptr);
  lstm_gemv_kernel<<<HH / 256, 256, 0, stream>>>(ctx, ENCD, h3, HH, nullptr, K4, b4, h4, out_h4);
  logits_kernel<<<(VOCAB + 255) / 256, 256, 0, stream>>>(h4, fck, fcb, out_logits);
}
